// Model_39676907884185
// MI455X (gfx1250) — matrix operand order repaired and checked by reading, not compiled, not run on hardware
//
#include <hip/hip_runtime.h>

#define BATCH 4
#define SEQ   4096
#define DIN   64
#define DOUT  128
#define PSTRIDE 80   // halves per LDS P-tile row (64 data + 16 pad, 160B: 16B aligned)
#define KSPLIT  4    // waves per block, each owning SEQ/KSPLIT keys of one q-tile

// scores/16 folded into q, plus log2(e) so v_exp_f32 (2^x) is the only exp op
#define QSCALE 0.09016844005556021f   // log2(e) / 16

typedef __attribute__((ext_vector_type(16))) _Float16 v16h;
typedef __attribute__((ext_vector_type(8)))  _Float16 v8h;
typedef __attribute__((ext_vector_type(8)))  float    v8f;
typedef __attribute__((ext_vector_type(4)))  float    v4f;

static __device__ __forceinline__ v16h cat16(v8h lo, v8h hi) {
    union { v16h v; v8h h[2]; } u;
    u.h[0] = lo; u.h[1] = hi;
    return u.v;
}

// ---------------------------------------------------------------------------
// Kernel 1: fused QKV projection.  One block per (b,n) row, 128 threads.
// q is pre-scaled by log2(e)/16 (score scaling folded in).  v stored
// transposed [B,128,N] so PV B-operands are contiguous 32B loads.
// ---------------------------------------------------------------------------
__global__ __launch_bounds__(128) void proj_kernel(
    const float* __restrict__ Q, const float* __restrict__ K, const float* __restrict__ V,
    const float* __restrict__ Wq, const float* __restrict__ bq,
    const float* __restrict__ Wk, const float* __restrict__ bk,
    const float* __restrict__ Wv, const float* __restrict__ bv,
    _Float16* __restrict__ qh, _Float16* __restrict__ kh, _Float16* __restrict__ vT)
{
    const int row = blockIdx.x;            // b*SEQ + n
    const int b   = row >> 12;
    const int n   = row & (SEQ - 1);
    const int t   = threadIdx.x;           // 0..127

    __shared__ float sq[DIN], sk[DIN], sv[DIN];
    if (t < DIN) {
        sq[t] = Q[(size_t)row * DIN + t];
        sk[t] = K[(size_t)row * DIN + t];
        sv[t] = V[(size_t)row * DIN + t];
    }
    __syncthreads();

    const int o = t;                       // output feature 0..127
    const v4f* wq = (const v4f*)(Wq + (size_t)o * DIN);
    const v4f* wk = (const v4f*)(Wk + (size_t)o * DIN);
    const v4f* wv = (const v4f*)(Wv + (size_t)o * DIN);
    const v4f* xq = (const v4f*)sq;
    const v4f* xk = (const v4f*)sk;
    const v4f* xv = (const v4f*)sv;

    float aq = bq[o], ak = bk[o], av = bv[o];
    #pragma unroll
    for (int d = 0; d < DIN / 4; ++d) {
        v4f q4 = xq[d], k4 = xk[d], v4 = xv[d];
        v4f a4 = wq[d], b4 = wk[d], c4 = wv[d];
        #pragma unroll
        for (int j = 0; j < 4; ++j) {
            aq = fmaf(q4[j], a4[j], aq);
            ak = fmaf(k4[j], b4[j], ak);
            av = fmaf(v4[j], c4[j], av);
        }
    }
    qh[(size_t)row * DOUT + o] = (_Float16)(aq * QSCALE);
    kh[(size_t)row * DOUT + o] = (_Float16)ak;
    vT[((size_t)b * DOUT + o) * SEQ + n] = (_Float16)av;
}

// ---------------------------------------------------------------------------
// Kernel 2: flash attention, split-K over keys.  One block = one 16-query
// tile; each of the 4 waves owns a 1024-key quarter (16 x 64-key blocks),
// then partials are merged through LDS.  The in-loop P transpose uses a
// wave-private LDS tile: same-wave DS ops are in-order per the ISA, so only
// a compiler memory barrier is needed (no s_barrier in the hot loop).
// ---------------------------------------------------------------------------
__global__ __launch_bounds__(128) void attn_kernel(
    const _Float16* __restrict__ qh, const _Float16* __restrict__ kh,
    const _Float16* __restrict__ vT, float* __restrict__ out)
{
    const int lane = threadIdx.x & 31;
    const int wave = threadIdx.x >> 5;     // 0..3 = key-range quarter
    const int l    = lane & 15;            // column / row-within-half index
    const int hi   = lane >> 4;            // which 16-lane half
    const int b    = blockIdx.y;
    const int q0   = blockIdx.x * 16;

    const _Float16* qb = qh + (size_t)b * SEQ * DOUT;
    const _Float16* kb = kh + (size_t)b * SEQ * DOUT;
    const _Float16* vb = vT + (size_t)b * DOUT * SEQ;

    __shared__ _Float16 plds[KSPLIT][16 * PSTRIDE];  // wave-private P tiles
    __shared__ float mls0[KSPLIT][16];               // per-wave row maxes
    __shared__ float mls1[KSPLIT][16];               // per-wave scaled row sums
    __shared__ float accs[KSPLIT][16][DOUT];         // per-wave scaled partial O
    _Float16* pt = plds[wave];

    // q A-fragments (16 rows x 128 = 4 K-slices of 32), loaded once.
    v16h aQ[4];
    {
        const _Float16* qrow = qb + (size_t)(q0 + l) * DOUT;
        #pragma unroll
        for (int kk = 0; kk < 4; ++kk) {
            v8h lo  = *(const v8h*)(qrow + kk * 32 + hi * 16);
            v8h hi8 = *(const v8h*)(qrow + kk * 32 + hi * 16 + 8);
            aQ[kk] = cat16(lo, hi8);
        }
    }

    // all-ones B fragment for row-sum WMMA
    v16h onesB;
    #pragma unroll
    for (int j = 0; j < 16; ++j) onesB[j] = (_Float16)1.0f;

    v8f acc[8] = {};                       // 16x128 f32 partial output
    float m_i[8], l_i[8];                  // running max (log2 domain) / sum
    #pragma unroll
    for (int r = 0; r < 8; ++r) { m_i[r] = -3.0e38f; l_i[r] = 0.0f; }

    const int keybeg = wave * (SEQ / KSPLIT);
    const int keyend = keybeg + (SEQ / KSPLIT);

    for (int key0 = keybeg; key0 < keyend; key0 += 64) {
        // ---- S tiles: tile t covers keys key0 + 4*n + t (col n) ----------
        v8f s[4] = {};
        const _Float16* kbase = kb + (size_t)(key0 + 4 * l) * DOUT + hi * 16;
        #pragma unroll
        for (int kk = 0; kk < 4; ++kk) {
            #pragma unroll
            for (int t = 0; t < 4; ++t) {
                v16h bkf = *(const v16h*)(kbase + (size_t)t * DOUT + kk * 32);
                s[t] = __builtin_amdgcn_wmma_f32_16x16x32_f16(false, aQ[kk], false, bkf,
                                                              (short)0, s[t], false, false);
            }
        }

        // ---- online softmax (exp2 domain; scaling already folded into q) -
        float alpha[8];
        #pragma unroll
        for (int r = 0; r < 8; ++r) {
            float x0 = s[0][r], x1 = s[1][r], x2 = s[2][r], x3 = s[3][r];
            float mx = fmaxf(fmaxf(x0, x1), fmaxf(x2, x3));
            mx = fmaxf(mx, __shfl_xor(mx, 1, 32));
            mx = fmaxf(mx, __shfl_xor(mx, 2, 32));
            mx = fmaxf(mx, __shfl_xor(mx, 4, 32));
            mx = fmaxf(mx, __shfl_xor(mx, 8, 32));   // stays inside each 16-lane half
            float mnew = fmaxf(m_i[r], mx);
            float a  = __builtin_amdgcn_exp2f(m_i[r] - mnew);
            float p0 = __builtin_amdgcn_exp2f(x0 - mnew);
            float p1 = __builtin_amdgcn_exp2f(x1 - mnew);
            float p2 = __builtin_amdgcn_exp2f(x2 - mnew);
            float p3 = __builtin_amdgcn_exp2f(x3 - mnew);
            m_i[r]   = mnew;
            alpha[r] = a;
            // row m = r + hi*8; keys 4l..4l+3 are adjacent cols -> one b64
            union { _Float16 h[4]; unsigned long long u; } pk;
            pk.h[0] = (_Float16)p0; pk.h[1] = (_Float16)p1;
            pk.h[2] = (_Float16)p2; pk.h[3] = (_Float16)p3;
            *(unsigned long long*)(pt + (r + hi * 8) * PSTRIDE + 4 * l) = pk.u;
        }

        // rescale running output accumulators (same row mapping as D layout)
        #pragma unroll
        for (int c = 0; c < 8; ++c)
            #pragma unroll
            for (int r = 0; r < 8; ++r)
                acc[c][r] *= alpha[r];

        // wave-private LDS tile + same-wave in-order DS: compiler fence only
        asm volatile("" ::: "memory");

        // ---- reload P in A layout: two K-slices of 32 ----
        const _Float16* prow = pt + l * PSTRIDE;
        v16h aP0 = cat16(*(const v8h*)(prow + hi * 16),
                         *(const v8h*)(prow + hi * 16 + 8));
        v16h aP1 = cat16(*(const v8h*)(prow + 32 + hi * 16),
                         *(const v8h*)(prow + 40 + hi * 16));

        // ---- row sums via P x ones (no butterfly shuffles needed) ----
        v8f srow = {};
        srow = __builtin_amdgcn_wmma_f32_16x16x32_f16(false, aP0, false, onesB,
                                                      (short)0, srow, false, false);
        srow = __builtin_amdgcn_wmma_f32_16x16x32_f16(false, aP1, false, onesB,
                                                      (short)0, srow, false, false);
        #pragma unroll
        for (int r = 0; r < 8; ++r)
            l_i[r] = l_i[r] * alpha[r] + srow[r];

        // ---- acc += P . V  (K-dim = keys; V rows in natural key order) ----
        #pragma unroll
        for (int c = 0; c < 8; ++c) {
            const _Float16* vrow = vb + (size_t)(c * 16 + l) * SEQ + key0 + hi * 16;
            v16h bv0 = *(const v16h*)(vrow);
            acc[c] = __builtin_amdgcn_wmma_f32_16x16x32_f16(false, aP0, false, bv0,
                                                            (short)0, acc[c], false, false);
            v16h bv1 = *(const v16h*)(vrow + 32);
            acc[c] = __builtin_amdgcn_wmma_f32_16x16x32_f16(false, aP1, false, bv1,
                                                            (short)0, acc[c], false, false);
        }
        asm volatile("" ::: "memory");
    }

    // ---- split-K merge across the 4 waves -------------------------------
    // 1) publish per-wave row maxes
    if (l == 0) {
        #pragma unroll
        for (int r = 0; r < 8; ++r)
            mls0[wave][r + hi * 8] = m_i[r];
    }
    __syncthreads();

    // 2) global row max -> per-wave rescale factor; publish scaled l and acc
    float fac[8];
    #pragma unroll
    for (int r = 0; r < 8; ++r) {
        int m = r + hi * 8;
        float g = fmaxf(fmaxf(mls0[0][m], mls0[1][m]),
                        fmaxf(mls0[2][m], mls0[3][m]));
        fac[r] = __builtin_amdgcn_exp2f(m_i[r] - g);
    }
    if (l == 0) {
        #pragma unroll
        for (int r = 0; r < 8; ++r)
            mls1[wave][r + hi * 8] = l_i[r] * fac[r];
    }
    #pragma unroll
    for (int r = 0; r < 8; ++r) {
        int m = r + hi * 8;
        #pragma unroll
        for (int c = 0; c < 8; ++c)
            accs[wave][m][c * 16 + l] = acc[c][r] * fac[r];
    }
    __syncthreads();

    // 3) final sum + normalize; thread = output column -> coalesced stores
    const int col = threadIdx.x;           // 0..127
    float* obase = out + ((size_t)b * SEQ + q0) * DOUT + col;
    #pragma unroll
    for (int m = 0; m < 16; ++m) {
        float ssum = accs[0][m][col] + accs[1][m][col]
                   + accs[2][m][col] + accs[3][m][col];
        float ltot = mls1[0][m] + mls1[1][m] + mls1[2][m] + mls1[3][m];
        obase[(size_t)m * DOUT] = ssum * (0.1f / ltot);
    }
}

// ---------------------------------------------------------------------------
extern "C" void kernel_launch(void* const* d_in, const int* in_sizes, int n_in,
                              void* d_out, int out_size, void* d_ws, size_t ws_size,
                              hipStream_t stream) {
    const float* Q  = (const float*)d_in[0];
    const float* K  = (const float*)d_in[1];
    const float* V  = (const float*)d_in[2];
    const float* Wq = (const float*)d_in[3];
    const float* bq = (const float*)d_in[4];
    const float* Wk = (const float*)d_in[5];
    const float* bk = (const float*)d_in[6];
    const float* Wv = (const float*)d_in[7];
    const float* bv = (const float*)d_in[8];
    float* out = (float*)d_out;

    const size_t elems = (size_t)BATCH * SEQ * DOUT;   // 2 Mi halves (4 MiB) each
    _Float16* qh = (_Float16*)d_ws;
    _Float16* kh = qh + elems;
    _Float16* vT = kh + elems;

    proj_kernel<<<dim3(BATCH * SEQ), dim3(128), 0, stream>>>(
        Q, K, V, Wq, bq, Wk, bk, Wv, bv, qh, kh, vT);

    attn_kernel<<<dim3(SEQ / 16, BATCH), dim3(128), 0, stream>>>(qh, kh, vT, out);
}
